// SecondOrderInteraction_24352464568558
// MI455X (gfx1250) — matrix operand order repaired and checked by reading, not compiled, not run on hardware
//
#include <hip/hip_runtime.h>

// FM second-order interaction for gfx1250 (MI455X).
// res[b] = 0.5 * ( ||x_b @ V||^2  -  sum_f x[b,f]^2 * w[f] ),  w[f] = sum_k V[f,k]^2
// B=16384, F=2048, K=64.
//
// Roofline: one 128 MiB streaming pass over x -> ~5.8 us @ 23.3 TB/s (x fits
// in the 192 MB L2 across timing replays). xv GEMM on v_wmma_f32_16x16x32_f16
// (f16 in, f32 accumulate); the x2v2 term is an exact-f32 matvec fused into
// the same x pass, accumulated in 4 independent v4f lanes (no per-iteration
// horizontal reduction / serial add chain). 2 waves per 16-row tile (F split
// in half, partial C tiles combined through LDS) -> 2048 waves chip-wide.

typedef __attribute__((ext_vector_type(16))) _Float16 v16h;
typedef __attribute__((ext_vector_type(8)))  float    v8f;
typedef __attribute__((ext_vector_type(4)))  float    v4f;

#define FM_B 16384
#define FM_F 2048
#define FM_K 64

// ---------------------------------------------------------------------------
// Prep 1: vt[n][f] = (f16) v[f][n]   (K x F transpose, 256 KB, L2-resident)
// ---------------------------------------------------------------------------
__global__ void fm_prep_vt(const float* __restrict__ v, _Float16* __restrict__ vt) {
  int tid = blockIdx.x * blockDim.x + threadIdx.x;   // 0 .. F*K-1
  int f = tid >> 6;
  int n = tid & 63;
  vt[(size_t)n * FM_F + f] = (_Float16)v[(size_t)f * FM_K + n];
}

// ---------------------------------------------------------------------------
// Prep 2: w[f] = sum_n v[f][n]^2   (2048 f32, 8 KB)
// ---------------------------------------------------------------------------
__global__ void fm_prep_w(const float* __restrict__ v, float* __restrict__ w) {
  int f = blockIdx.x * blockDim.x + threadIdx.x;     // 0 .. F-1
  const float* vf = v + (size_t)f * FM_K;
  float s = 0.0f;
#pragma unroll
  for (int n = 0; n < FM_K; n += 4) {
    v4f t = *(const v4f*)(vf + n);
    s += t.x * t.x + t.y * t.y + t.z * t.z + t.w * t.w;
  }
  w[f] = s;
}

// ---------------------------------------------------------------------------
// Main: block = 64 threads = 2 waves sharing one 16-row tile (F split 2-ways).
// Per 32-wide F step per wave: 4x v_wmma_f32_16x16x32_f16 + packed-f32 matvec.
// ---------------------------------------------------------------------------
__global__ void __launch_bounds__(64)
fm_main(const float* __restrict__ x,
        const _Float16* __restrict__ vt,
        const float* __restrict__ w,
        float* __restrict__ out) {
  __shared__ float lds_acc[32][36];   // wave 1's partial C tiles (padded rows)
  __shared__ float lds_s2[2][16];     // per-wave, per-row x2*w partial sums

  const int lane = threadIdx.x & 31;
  const int wid  = threadIdx.x >> 5;          // which F-half this wave owns
  const int m    = lane & 15;                 // row for A, col for B/C
  const int hi   = lane >> 4;                 // half-wave select
  const int row0 = blockIdx.x * 16;           // block's 16-row tile
  const int arow = row0 + m;

  const float* xrow  = x + (size_t)arow * FM_F;
  const int selA  = hi * 8;                   // A-frag K-run start in 32-chunk
  const int fbase = wid * (FM_F / 2);
  const int fend  = fbase + (FM_F / 2);

  v8f acc[4] = {};                            // xv tiles, n = 16t + m
  v4f s2v0 = {}, s2v1 = {}, s2v2 = {}, s2v3 = {};   // independent x^2*w accs

  for (int kb = fbase; kb < fend; kb += 32) {
    // ---- A fragment (16-bit A 16x32 layout) + x^2*w matvec off same data ----
    // lane<16 holds K = {0..7, 16..23}; lane>=16 holds K = {8..15, 24..31}
    v4f x0 = *(const v4f*)(xrow + kb + selA);
    v4f x1 = *(const v4f*)(xrow + kb + selA + 4);
    v4f x2 = *(const v4f*)(xrow + kb + 16 + selA);
    v4f x3 = *(const v4f*)(xrow + kb + 16 + selA + 4);
    v4f w0 = *(const v4f*)(w + kb + selA);
    v4f w1 = *(const v4f*)(w + kb + selA + 4);
    v4f w2 = *(const v4f*)(w + kb + 16 + selA);
    v4f w3 = *(const v4f*)(w + kb + 16 + selA + 4);

    v16h a;
#pragma unroll
    for (int i = 0; i < 4; ++i) {
      a[i]      = (_Float16)x0[i];
      a[4 + i]  = (_Float16)x1[i];
      a[8 + i]  = (_Float16)x2[i];
      a[12 + i] = (_Float16)x3[i];
    }

    // Exact-f32 matvec: elementwise packed FMAs, no cross-lane/horizontal ops
    // in the loop, four independent recurrences.
    s2v0 += (x0 * x0) * w0;
    s2v1 += (x1 * x1) * w1;
    s2v2 += (x2 * x2) * w2;
    s2v3 += (x3 * x3) * w3;

    // speculative stream prefetch of x (global_prefetch_b8)
    __builtin_prefetch(xrow + kb + 256, 0, 0);

    const int kB = kb + hi * 8;               // B frag: K = {8h..8h+7, 16+8h..16+8h+7}
#pragma unroll
    for (int t = 0; t < 4; ++t) {
      const _Float16* bp = vt + (size_t)(t * 16 + m) * FM_F + kB;
      typedef __attribute__((ext_vector_type(8))) _Float16 v8h_;
      v8h_ b0 = *(const v8h_*)bp;
      v8h_ b1 = *(const v8h_*)(bp + 16);
      v16h b = __builtin_shufflevector(b0, b1, 0, 1, 2, 3, 4, 5, 6, 7, 8, 9, 10, 11, 12, 13, 14, 15);  // 2x global_load_b128, L2 hit
      acc[t] = __builtin_amdgcn_wmma_f32_16x16x32_f16(
          false, a, false, b, (short)0, acc[t], false, false);
    }
  }

  // ---- horizontal reduce of the matvec accumulators (once) ----
  v4f sv = (s2v0 + s2v1) + (s2v2 + s2v3);
  float s2 = (sv.x + sv.y) + (sv.z + sv.w);

  // ---- combine the two F-halves through LDS ----
  // s2 per full row (this wave's F-half): lanes m and m+16 both hold row m.
  float s2p = s2 + __shfl_xor(s2, 16, 32);
  if (wid == 1) {
#pragma unroll
    for (int t = 0; t < 4; ++t)
      *(v8f*)&lds_acc[lane][t * 8] = acc[t];  // 2x ds_store_b128 each
  }
  if (hi == 0) lds_s2[wid][m] = s2p;
  __syncthreads();

  if (wid == 0) {
#pragma unroll
    for (int t = 0; t < 4; ++t)
      acc[t] += *(const v8f*)&lds_acc[lane][t * 8];

    // C layout: lane holds col n=m; VGPR r holds row M = hi*8 + r.
#pragma unroll
    for (int r = 0; r < 8; ++r) {
      float s1 = acc[0][r] * acc[0][r] + acc[1][r] * acc[1][r] +
                 acc[2][r] * acc[2][r] + acc[3][r] * acc[3][r];
#pragma unroll
      for (int off = 1; off < 16; off <<= 1) {  // stays within 16-lane halves
        s1 += __shfl_xor(s1, off, 32);
      }
      if (m == 0) {
        const int rr = hi * 8 + r;
        out[row0 + rr] = 0.5f * (s1 - (lds_s2[0][rr] + lds_s2[1][rr]));
      }
    }
  }
}

// ---------------------------------------------------------------------------
extern "C" void kernel_launch(void* const* d_in, const int* in_sizes, int n_in,
                              void* d_out, int out_size, void* d_ws, size_t ws_size,
                              hipStream_t stream) {
  const float* x = (const float*)d_in[0];   // (B, F) f32
  const float* v = (const float*)d_in[1];   // (F, K) f32
  float* out     = (float*)d_out;           // (B,)  f32

  _Float16* vt = (_Float16*)d_ws;                                              // 256 KB
  float*    w  = (float*)((char*)d_ws + (size_t)FM_F * FM_K * sizeof(_Float16)); // 8 KB

  fm_prep_vt<<<(FM_F * FM_K) / 256, 256, 0, stream>>>(v, vt);
  fm_prep_w<<<FM_F / 256, 256, 0, stream>>>(v, w);
  fm_main<<<FM_B / 16, 64, 0, stream>>>(x, vt, w, out);
}
